// BertLayer_71794673320326
// MI455X (gfx1250) — compile-verified, matrix operand order checked by reading, not run on hardware
//
#include <hip/hip_runtime.h>
#include <hip/hip_bf16.h>

// ---------------- problem constants (fixed by the reference) ----------------
#define Bn   8
#define Sn   512
#define Hn   1024
#define NHn  16
#define Dn   64          // head dim
#define FFn  4096
#define Rn   (Bn*Sn)     // 4096 token rows
#define BHn  (Bn*NHn)    // 128 (batch, head) pairs
#define CHUNK 16         // heads per attention chunk (caps scores buffer)
#define NCHUNK (BHn/CHUNK)

typedef __attribute__((ext_vector_type(16))) __bf16 v16bf;
typedef __attribute__((ext_vector_type(8)))  float  v8f;

enum { EPI_QKV = 1, EPI_VT = 2, EPI_SCORES = 3, EPI_CTX = 4, EPI_BIASRES = 5, EPI_GELU = 6 };

__device__ __forceinline__ float gelu_f(float x) {
    const float c = 0.7978845608028654f; // sqrt(2/pi)
    float t = c * (x + 0.044715f * x * x * x);
    return x * 0.5f * (1.0f + tanhf(t));
}

// async 16B global -> LDS copy (per-lane), tracked by ASYNCcnt
__device__ __forceinline__ void async_cp16(unsigned lds_byte_addr, const void* gaddr) {
    asm volatile("global_load_async_to_lds_b128 %0, %1, off"
                 :: "v"(lds_byte_addr), "v"((unsigned long long)(uintptr_t)gaddr)
                 : "memory");
}

// ---------------- f32 -> bf16 conversion (row-major copy) ----------------
__global__ __launch_bounds__(256)
void cvt_f32_bf16(const float* __restrict__ in, __bf16* __restrict__ out, long n) {
    long i = ((long)blockIdx.x * 256 + threadIdx.x) * 4;
    if (i + 3 < n) {
        float4 v = *(const float4*)(in + i);
        __bf16 o[4] = {(__bf16)v.x, (__bf16)v.y, (__bf16)v.z, (__bf16)v.w};
        *(uint2*)(out + i) = *(const uint2*)o;
    } else {
        for (; i < n; ++i) out[i] = (__bf16)in[i];
    }
}

// ---------------- f32 [Kr][Nc] -> bf16 [Nc][Kr] transpose (32x32 tiles) -----
__global__ __launch_bounds__(256)
void transpose_f32_bf16(const float* __restrict__ in, __bf16* __restrict__ out,
                        int Kr, int Nc) {
    __shared__ float tile[32][33];
    const int bx = blockIdx.x * 32;           // along Nc
    const int by = blockIdx.y * 32;           // along Kr
    const int tx = threadIdx.x & 31, ty = threadIdx.x >> 5;  // 8 rows per pass
    #pragma unroll
    for (int i = 0; i < 32; i += 8)
        tile[ty + i][tx] = in[(long)(by + ty + i) * Nc + (bx + tx)];
    __syncthreads();
    #pragma unroll
    for (int i = 0; i < 32; i += 8)
        out[(long)(bx + ty + i) * Kr + (by + tx)] = (__bf16)tile[tx][ty + i];
}

// ---------------- bf16 GEMM core: v_wmma_f32_16x16x32_bf16 ------------------
// C[M,N] = A[M,K] @ B[K,N] with B supplied TRANSPOSED: (k,n) = Bt[n*ldb + k].
// All staging is async b128 global->LDS, double buffered; KB=64 (2 WMMA
// K-slices, 16 WMMAs) per barrier pair; steady-state loop is branch-free.
template<int BM, int BN, int WMT, int WNT, int EPI>
__global__ __launch_bounds__(256)
void gemm_bf16_k(const __bf16* __restrict__ A, long lda, long asz,
                 const __bf16* __restrict__ Bt, long ldb, long bsz,
                 int M, int N, int K,
                 const float* __restrict__ bias, const float* __restrict__ resid,
                 float* __restrict__ outf, __bf16* __restrict__ outb, long osz,
                 const int* __restrict__ mask, float scale, int head_base)
{
    constexpr int KB   = 64;                   // two WMMA K-slices per stage
    constexpr int ROWE = KB + 8;               // 72 elems = 144B rows (16B aligned)
    constexpr int WAVES_M = BM / WMT;
    constexpr int TM = WMT / 16, TN = WNT / 16;
    constexpr int NA = BM * 8 / 256;           // async b128 copies per thread (A)
    constexpr int NB = BN * 8 / 256;           // async b128 copies per thread (B)
    constexpr int CPT = NA + NB;               // ASYNCcnt per wave per stage
    constexpr unsigned ABUF = BM * ROWE * 2;   // bytes per A buffer
    constexpr unsigned BBUF = BN * ROWE * 2;

    __shared__ __align__(16) __bf16 As[2][BM][ROWE];
    __shared__ __align__(16) __bf16 Bs[2][BN][ROWE];

    const int tid   = threadIdx.x;
    const int wave  = tid >> 5, lane = tid & 31;
    const int lhalf = lane >> 4, lrow = lane & 15;
    const int wm    = wave % WAVES_M, wn = wave / WAVES_M;
    const int row0  = blockIdx.x * BM, col0 = blockIdx.y * BN;
    const int hz    = blockIdx.z;

    A  += (long)hz * asz;
    Bt += (long)hz * bsz;
    float* outfz = outf ? (outf + (long)hz * osz) : nullptr;

    // truncation of generic shared pointer gives LDS byte offset (ISA aperture rule)
    const unsigned asb = (unsigned)(uintptr_t)&As[0][0][0];
    const unsigned bsb = (unsigned)(uintptr_t)&Bs[0][0][0];

    v8f acc[TM][TN];
    #pragma unroll
    for (int i = 0; i < TM; ++i)
        #pragma unroll
        for (int j = 0; j < TN; ++j)
            #pragma unroll
            for (int g = 0; g < 8; ++g) acc[i][j][g] = 0.0f;

    auto stage = [&](int k0, int buf) {
        #pragma unroll
        for (int u = 0; u < NA; ++u) {
            const int it = tid + u * 256;
            const int r = it >> 3, kg = (it & 7) << 3;
            async_cp16(asb + buf * ABUF + (unsigned)(r * ROWE + kg) * 2,
                       A + (long)(row0 + r) * lda + (k0 + kg));
        }
        #pragma unroll
        for (int u = 0; u < NB; ++u) {
            const int it = tid + u * 256;
            const int nn = it >> 3, kg = (it & 7) << 3;
            async_cp16(bsb + buf * BBUF + (unsigned)(nn * ROWE + kg) * 2,
                       Bt + (long)(col0 + nn) * ldb + (k0 + kg));
        }
    };

    // fragments per ISA 16-bit 16x32 layout:
    // lane<16: elems 0..7 -> K 0..7, elems 8..15 -> K 16..23; lane>=16: +8
    auto compute = [&](int buf) {
        #pragma unroll
        for (int ks = 0; ks < KB / 32; ++ks) {
            v16bf afr[TM], bfr[TN];
            #pragma unroll
            for (int i = 0; i < TM; ++i) {
                const __bf16* p = &As[buf][wm * WMT + i * 16 + lrow][ks * 32 + lhalf * 8];
                ((uint4*)&afr[i])[0] = *(const uint4*)p;
                ((uint4*)&afr[i])[1] = *(const uint4*)(p + 16);
            }
            #pragma unroll
            for (int j = 0; j < TN; ++j) {
                const __bf16* p = &Bs[buf][wn * WNT + j * 16 + lrow][ks * 32 + lhalf * 8];
                ((uint4*)&bfr[j])[0] = *(const uint4*)p;
                ((uint4*)&bfr[j])[1] = *(const uint4*)(p + 16);
            }
            #pragma unroll
            for (int i = 0; i < TM; ++i)
                #pragma unroll
                for (int j = 0; j < TN; ++j)
                    acc[i][j] = __builtin_amdgcn_wmma_f32_16x16x32_bf16(
                        false, afr[i], false, bfr[j], (short)0, acc[i][j], false, false);
        }
    };

    const int nk = K / KB;
    stage(0, 0);
    // steady state: branch-free (stage next, wait for previous batch, compute)
    for (int t = 0; t < nk - 1; ++t) {
        stage((t + 1) * KB, (t + 1) & 1);
        asm volatile("s_wait_asynccnt %0" :: "i"(CPT) : "memory");  // tile t landed
        __syncthreads();
        compute(t & 1);
        __syncthreads();                        // buffer free for overwrite at t+2
    }
    // peeled last tile
    asm volatile("s_wait_asynccnt 0" ::: "memory");
    __syncthreads();
    compute((nk - 1) & 1);

    // ---- epilogue: C layout lane holds (M = g + 8*lhalf, N = lrow) ----
    const int head = head_base + hz;
    for (int i = 0; i < TM; ++i) {
        for (int j = 0; j < TN; ++j) {
            const int rbase = row0 + wm * WMT + i * 16 + 8 * lhalf;
            const int c     = col0 + wn * WNT + j * 16 + lrow;
            if (EPI == EPI_VT) {                // bias + scatter to [B*NH][D][S], packed
                const float bv_ = bias[c];
                const int bb = rbase / Sn, s0 = rbase % Sn;   // 8-aligned, same batch row
                const int n = c / Dn, d = c % Dn;
                __bf16 pk[8];
                #pragma unroll
                for (int g = 0; g < 8; ++g) pk[g] = (__bf16)(acc[i][j][g] + bv_);
                *(uint4*)(outb + (((long)(bb * NHn + n)) * Dn + d) * Sn + s0) = *(const uint4*)pk;
                continue;
            }
            #pragma unroll
            for (int g = 0; g < 8; ++g) {
                const int r = rbase + g;
                float v = acc[i][j][g];
                if (EPI == EPI_QKV) {           // bias + scatter to [B*NH][S][D]
                    v += bias[c];
                    int bb = r / Sn, s = r % Sn, n = c / Dn, d = c % Dn;
                    outb[(((long)(bb * NHn + n)) * Sn + s) * Dn + d] = (__bf16)v;
                } else if (EPI == EPI_SCORES) { // 1/sqrt(D) + additive mask, f32
                    int bb = head / NHn;
                    v = v * scale + (1.0f - (float)mask[bb * Sn + c]) * -10000.0f;
                    outfz[(long)r * N + c] = v;
                } else if (EPI == EPI_CTX) {    // gather heads back to [B,S,H] bf16
                    int bb = head / NHn, n = head % NHn;
                    outb[((long)(bb * Sn + r)) * Hn + n * Dn + c] = (__bf16)v;
                } else if (EPI == EPI_BIASRES) {// + bias + residual, f32 (pre-LN)
                    outf[(long)r * N + c] = v + bias[c] + resid[(long)r * N + c];
                } else if (EPI == EPI_GELU) {   // gelu(bias + v) -> bf16
                    v += bias[c];
                    outb[(long)r * N + c] = (__bf16)gelu_f(v);
                }
            }
        }
    }
}

// ---------------- row softmax over S=512 f32, writes bf16 in place ----------
__global__ __launch_bounds__(128)
void softmax_bf16_inplace(float* __restrict__ scores)
{
    const long row = blockIdx.x;
    float* p = scores + row * (long)Sn;
    const int tid = threadIdx.x;
    float4 v = ((const float4*)p)[tid];          // 128 thr * 4 = 512
    __shared__ float red[128];

    red[tid] = fmaxf(fmaxf(v.x, v.y), fmaxf(v.z, v.w));
    __syncthreads();
    for (int s = 64; s > 0; s >>= 1) { if (tid < s) red[tid] = fmaxf(red[tid], red[tid + s]); __syncthreads(); }
    const float rmax = red[0];
    __syncthreads();

    float e0 = __expf(v.x - rmax), e1 = __expf(v.y - rmax);
    float e2 = __expf(v.z - rmax), e3 = __expf(v.w - rmax);
    red[tid] = e0 + e1 + e2 + e3;
    __syncthreads();
    for (int s = 64; s > 0; s >>= 1) { if (tid < s) red[tid] += red[tid + s]; __syncthreads(); }
    const float inv = 1.0f / red[0];

    __bf16 o[4] = {(__bf16)(e0 * inv), (__bf16)(e1 * inv), (__bf16)(e2 * inv), (__bf16)(e3 * inv)};
    ((uint2*)p)[tid] = *(const uint2*)o;         // bf16 row in first half of f32 row
}

// ---------------- LayerNorm over H=1024, f32 out (+ optional bf16 copy) -----
__global__ __launch_bounds__(256)
void layernorm_k(const float* __restrict__ in, const float* __restrict__ g,
                 const float* __restrict__ beta, float* __restrict__ outf,
                 __bf16* __restrict__ outb)
{
    const long row = blockIdx.x;
    const int tid = threadIdx.x;
    const float4 v = ((const float4*)(in + row * Hn))[tid];
    __shared__ float red[256];

    red[tid] = v.x + v.y + v.z + v.w;
    __syncthreads();
    for (int s = 128; s > 0; s >>= 1) { if (tid < s) red[tid] += red[tid + s]; __syncthreads(); }
    const float mu = red[0] * (1.0f / Hn);
    __syncthreads();

    const float dx = v.x - mu, dy = v.y - mu, dz = v.z - mu, dw = v.w - mu;
    red[tid] = dx * dx + dy * dy + dz * dz + dw * dw;
    __syncthreads();
    for (int s = 128; s > 0; s >>= 1) { if (tid < s) red[tid] += red[tid + s]; __syncthreads(); }
    const float inv = rsqrtf(red[0] * (1.0f / Hn) + 1e-5f);

    const float4 gg = ((const float4*)g)[tid];
    const float4 bb = ((const float4*)beta)[tid];
    const float y0 = dx * inv * gg.x + bb.x;
    const float y1 = dy * inv * gg.y + bb.y;
    const float y2 = dz * inv * gg.z + bb.z;
    const float y3 = dw * inv * gg.w + bb.w;
    float4 yo = {y0, y1, y2, y3};
    ((float4*)(outf + row * Hn))[tid] = yo;
    if (outb) {
        __bf16 ob[4] = {(__bf16)y0, (__bf16)y1, (__bf16)y2, (__bf16)y3};
        ((uint2*)(outb + row * Hn))[tid] = *(const uint2*)ob;
    }
}

// ---------------- host orchestration ----------------
extern "C" void kernel_launch(void* const* d_in, const int* in_sizes, int n_in,
                              void* d_out, int out_size, void* d_ws, size_t ws_size,
                              hipStream_t stream)
{
    const float* x   = (const float*)d_in[0];
    const float* Wq  = (const float*)d_in[1];  const float* bq  = (const float*)d_in[2];
    const float* Wk  = (const float*)d_in[3];  const float* bk  = (const float*)d_in[4];
    const float* Wv  = (const float*)d_in[5];  const float* bv  = (const float*)d_in[6];
    const float* Wo  = (const float*)d_in[7];  const float* bo  = (const float*)d_in[8];
    const float* g1  = (const float*)d_in[9];  const float* b1  = (const float*)d_in[10];
    const float* Wi  = (const float*)d_in[11]; const float* bi  = (const float*)d_in[12];
    const float* Wf  = (const float*)d_in[13]; const float* bf_ = (const float*)d_in[14];
    const float* g2  = (const float*)d_in[15]; const float* b2  = (const float*)d_in[16];
    const int*   am  = (const int*)d_in[17];
    float* out = (float*)d_out;

    // workspace carve-up (~170 MB total)
    char* w = (char*)d_ws;
    auto take = [&](size_t bytes) -> char* {
        char* p = w; w += (bytes + 255) & ~(size_t)255; return p;
    };
    __bf16* xb    = (__bf16*)take((size_t)Rn * Hn * 2);
    __bf16* WqT   = (__bf16*)take((size_t)Hn * Hn * 2);    // [N][K] transposed weights
    __bf16* WkT   = (__bf16*)take((size_t)Hn * Hn * 2);
    __bf16* WvT   = (__bf16*)take((size_t)Hn * Hn * 2);
    __bf16* WoT   = (__bf16*)take((size_t)Hn * Hn * 2);
    __bf16* WiT   = (__bf16*)take((size_t)Hn * FFn * 2);   // [FF][H]
    __bf16* WfT   = (__bf16*)take((size_t)FFn * Hn * 2);   // [H][FF]
    __bf16* qb    = (__bf16*)take((size_t)BHn * Sn * Dn * 2);  // [head][s][d]
    __bf16* kb    = (__bf16*)take((size_t)BHn * Sn * Dn * 2);  // [head][s][d]
    __bf16* vbt   = (__bf16*)take((size_t)BHn * Dn * Sn * 2);  // [head][d][s]
    __bf16* ctxb  = (__bf16*)take((size_t)Rn * Hn * 2);
    float*  pre1  = (float*)take((size_t)Rn * Hn * 4);
    float*  attnf = (float*)take((size_t)Rn * Hn * 4);
    __bf16* attnb = (__bf16*)take((size_t)Rn * Hn * 2);
    __bf16* inter = (__bf16*)take((size_t)Rn * FFn * 2);
    float*  pre2  = (float*)take((size_t)Rn * Hn * 4);
    float*  scbuf = (float*)take((size_t)CHUNK * Sn * Sn * 4);
    (void)ws_size; (void)in_sizes; (void)n_in; (void)out_size;

    cvt_f32_bf16<<<dim3((Rn * Hn) / 1024), dim3(256), 0, stream>>>(x, xb, (long)Rn * Hn);
    auto tr = [&](const float* src, __bf16* dst, int Kr, int Nc) {
        transpose_f32_bf16<<<dim3(Nc / 32, Kr / 32), dim3(256), 0, stream>>>(src, dst, Kr, Nc);
    };
    tr(Wq, WqT, Hn, Hn);   tr(Wk, WkT, Hn, Hn);
    tr(Wv, WvT, Hn, Hn);   tr(Wo, WoT, Hn, Hn);
    tr(Wi, WiT, Hn, FFn);  tr(Wf, WfT, FFn, Hn);

    const dim3 blk(256);

    // ---- QKV projections (q,k row layout; v transposed per head)
    const dim3 gQKV(Rn / 128, Hn / 128, 1);
    gemm_bf16_k<128,128,64,32,EPI_QKV><<<gQKV, blk, 0, stream>>>(
        xb, Hn, 0, WqT, Hn, 0, Rn, Hn, Hn, bq, nullptr, nullptr, qb, 0, nullptr, 0.f, 0);
    gemm_bf16_k<128,128,64,32,EPI_QKV><<<gQKV, blk, 0, stream>>>(
        xb, Hn, 0, WkT, Hn, 0, Rn, Hn, Hn, bk, nullptr, nullptr, kb, 0, nullptr, 0.f, 0);
    gemm_bf16_k<128,128,64,32,EPI_VT><<<gQKV, blk, 0, stream>>>(
        xb, Hn, 0, WvT, Hn, 0, Rn, Hn, Hn, bv, nullptr, nullptr, vbt, 0, nullptr, 0.f, 0);

    // ---- attention, 8 chunks of 16 heads
    for (int ch = 0; ch < NCHUNK; ++ch) {
        const long qkoff = (long)ch * CHUNK * Sn * Dn;
        // scores = (q @ k^T)/8 + mask : Bt = k ([s][d] is k-contiguous per col)
        gemm_bf16_k<128,128,64,32,EPI_SCORES><<<dim3(Sn/128, Sn/128, CHUNK), blk, 0, stream>>>(
            qb + qkoff, Dn, (long)Sn * Dn,
            kb + qkoff, Dn, (long)Sn * Dn,
            Sn, Sn, Dn, nullptr, nullptr,
            scbuf, nullptr, (long)Sn * Sn, am, 0.125f, ch * CHUNK);
        softmax_bf16_inplace<<<dim3(CHUNK * Sn), dim3(128), 0, stream>>>(scbuf);
        // ctx = probs @ v : Bt = v^T [d][s] (k-contiguous), probs rows contiguous
        gemm_bf16_k<128,64,32,32,EPI_CTX><<<dim3(Sn/128, 1, CHUNK), blk, 0, stream>>>(
            (const __bf16*)scbuf, 2L * Sn, 2L * Sn * Sn,
            vbt + (long)ch * CHUNK * Dn * Sn, Sn, (long)Dn * Sn,
            Sn, Dn, Sn, nullptr, nullptr,
            nullptr, ctxb, 0, nullptr, 0.f, ch * CHUNK);
    }

    // ---- O-projection + bo + residual(x) -> pre-LN f32
    gemm_bf16_k<128,128,64,32,EPI_BIASRES><<<dim3(Rn/128, Hn/128, 1), blk, 0, stream>>>(
        ctxb, Hn, 0, WoT, Hn, 0, Rn, Hn, Hn, bo, x, pre1, nullptr, 0, nullptr, 0.f, 0);
    layernorm_k<<<dim3(Rn), dim3(256), 0, stream>>>(pre1, g1, b1, attnf, attnb);

    // ---- FF1: gelu(attn @ Wi + bi) -> bf16
    gemm_bf16_k<128,128,64,32,EPI_GELU><<<dim3(Rn/128, FFn/128, 1), blk, 0, stream>>>(
        attnb, Hn, 0, WiT, Hn, 0, Rn, FFn, Hn, bi, nullptr, nullptr, inter, 0, nullptr, 0.f, 0);

    // ---- FF2: inter @ Wf + bf + residual(attn) -> pre-LN f32
    gemm_bf16_k<128,128,64,32,EPI_BIASRES><<<dim3(Rn/128, Hn/128, 1), blk, 0, stream>>>(
        inter, FFn, 0, WfT, FFn, 0, Rn, Hn, FFn, bf_, attnf, pre2, nullptr, 0, nullptr, 0.f, 0);
    layernorm_k<<<dim3(Rn), dim3(256), 0, stream>>>(pre2, g2, b2, out, nullptr);
}
